// Attention_91036126806408
// MI455X (gfx1250) — matrix operand order repaired and checked by reading, not compiled, not run on hardware
//
#include <hip/hip_runtime.h>

#define B_      2
#define SEQ     2048
#define HEADS_  16
#define DH      64
#define DMODEL  1024
#define QKVN    3072

typedef __attribute__((ext_vector_type(16))) __bf16 bf16x16;
typedef __attribute__((ext_vector_type(8)))  __bf16 bf16x8;
typedef __attribute__((ext_vector_type(8)))  float  v8f;
typedef int v4i_ __attribute__((vector_size(16)));

// -------- async global->LDS copy (CDNA5 ASYNCcnt path), guarded fallback ----
#if defined(__AMDGCN__) && \
    __has_builtin(__builtin_amdgcn_global_load_async_to_lds_b128) && \
    __has_builtin(__builtin_amdgcn_s_wait_asynccnt)
#define USE_ASYNC_LDS 1
#else
#define USE_ASYNC_LDS 0
#endif

__device__ inline void copy16_to_lds(const void* g, void* l) {
#if USE_ASYNC_LDS
  __builtin_amdgcn_global_load_async_to_lds_b128(
      (__attribute__((address_space(1))) v4i_*)g,
      (__attribute__((address_space(3))) v4i_*)l, 0, 0);
#else
  *(bf16x8*)l = *(const bf16x8*)g;
#endif
}

__device__ inline void async_wait() {
#if USE_ASYNC_LDS
  __builtin_amdgcn_s_wait_asynccnt(0);
#endif
}

__device__ inline bf16x16 mk16(bf16x8 lo, bf16x8 hi) {
  bf16x16 r;
#pragma unroll
  for (int i = 0; i < 8; ++i) { r[i] = lo[i]; r[i + 8] = hi[i]; }
  return r;
}

__device__ inline v8f wmma_bf16(bf16x16 a, bf16x16 b, v8f c) {
  return __builtin_amdgcn_wmma_f32_16x16x32_bf16(false, a, false, b, (short)0, c,
                                                 false, false);
}

// ---------------------------------------------------------------------------
// Kernel 1: qkv = x @ w_qkv, fused RoPE on q,k; outputs bf16 in [b,h,n,dh]
// (fp32->bf16 conversion during staging, so this stays on the VALU copy path)
// ---------------------------------------------------------------------------
__global__ __launch_bounds__(256) void qkv_rope_kernel(
    const float* __restrict__ x, const float* __restrict__ w,
    __bf16* __restrict__ qbuf, __bf16* __restrict__ kbuf,
    __bf16* __restrict__ vbuf) {
  __shared__ __bf16 sA[128][32];   // rows x k
  __shared__ __bf16 sB[128][32];   // cols x k (transposed)

  const int tid  = threadIdx.x;
  const int wave = tid >> 5, lane = tid & 31;
  const int wm = wave >> 1;        // 0..3 : 32-row band
  const int wn = wave & 1;         // 0..1 : 64-col band
  const int tileM = blockIdx.x * 128;
  const int tileN = blockIdx.y * 128;
  const int l15  = lane & 15;
  const int ah   = (lane < 16) ? 0 : 16;   // A-frag k-offset
  const int bhf  = (lane < 16) ? 0 : 16;   // B-frag k-offset

  const v8f vzero = {0.f, 0.f, 0.f, 0.f, 0.f, 0.f, 0.f, 0.f};
  v8f acc[2][4];
#pragma unroll
  for (int i = 0; i < 2; ++i)
#pragma unroll
    for (int j = 0; j < 4; ++j) acc[i][j] = vzero;

  for (int k0 = 0; k0 < DMODEL; k0 += 32) {
    // A tile: 128x32 fp32 -> bf16
#pragma unroll
    for (int p = 0; p < 4; ++p) {
      int r = p * 32 + (tid >> 3);
      int c = (tid & 7) * 4;
      float4 f = *(const float4*)(x + (size_t)(tileM + r) * DMODEL + k0 + c);
      sA[r][c + 0] = (__bf16)f.x; sA[r][c + 1] = (__bf16)f.y;
      sA[r][c + 2] = (__bf16)f.z; sA[r][c + 3] = (__bf16)f.w;
    }
    // B tile: 32x128 fp32 -> bf16 transposed to [col][k]
#pragma unroll
    for (int p = 0; p < 4; ++p) {
      int r = p * 8 + (tid >> 5);
      int c = (tid & 31) * 4;
      float4 f = *(const float4*)(w + (size_t)(k0 + r) * QKVN + tileN + c);
      sB[c + 0][r] = (__bf16)f.x; sB[c + 1][r] = (__bf16)f.y;
      sB[c + 2][r] = (__bf16)f.z; sB[c + 3][r] = (__bf16)f.w;
    }
    __syncthreads();

    bf16x16 af[2], bf[4];
#pragma unroll
    for (int mt = 0; mt < 2; ++mt) {
      int row = wm * 32 + mt * 16 + l15;
      af[mt] = mk16(*(const bf16x8*)&sA[row][ah],
                    *(const bf16x8*)&sA[row][ah + 8]);
    }
#pragma unroll
    for (int nt = 0; nt < 4; ++nt) {
      int col = wn * 64 + nt * 16 + l15;
      bf[nt] = *(const bf16x16*)&sB[col][bhf];
    }
#pragma unroll
    for (int mt = 0; mt < 2; ++mt)
#pragma unroll
      for (int nt = 0; nt < 4; ++nt)
        acc[mt][nt] = wmma_bf16(af[mt], bf[nt], acc[mt][nt]);
    __syncthreads();
  }

  // epilogue: split q/k/v, RoPE on q,k (partner tile nt^2 in-register)
  const int rofs = (lane < 16) ? 0 : 8;
#pragma unroll
  for (int mt = 0; mt < 2; ++mt) {
#pragma unroll
    for (int nt = 0; nt < 4; ++nt) {
#pragma unroll
      for (int r = 0; r < 8; ++r) {
        int grow = tileM + wm * 32 + mt * 16 + r + rofs;
        int gcol = tileN + wn * 64 + nt * 16 + l15;
        int b    = grow >> 11;
        int pos  = grow & (SEQ - 1);
        int sec  = gcol >> 10;          // 0=q 1=k 2=v
        int wi   = gcol & 1023;
        int head = wi >> 6;
        int dh   = wi & 63;
        float v  = acc[mt][nt][r];
        size_t o = (((size_t)b * HEADS_ + head) * SEQ + pos) * DH + dh;
        if (sec == 2) {
          vbuf[o] = (__bf16)v;
        } else {
          float pv = acc[mt][nt ^ 2][r];      // dh +/- 32 partner
          int jj = dh & 31;
          float inv_freq = __expf(-9.210340371976184f * (float)(2 * jj) *
                                  (1.0f / 64.0f));
          float sn, cs;
          __sincosf((float)pos * inv_freq, &sn, &cs);
          float rot = (dh < 32) ? -pv : pv;
          float outv = v * cs + rot * sn;
          (sec == 0 ? qbuf : kbuf)[o] = (__bf16)outv;
        }
      }
    }
  }
}

// ---------------------------------------------------------------------------
// Kernel 2: causal flash attention. block = (rowTile of 128) x (b*h).
// Q and K tiles staged via GLOBAL_LOAD_ASYNC_TO_LDS_B128 (ASYNCcnt path);
// V needs an in-flight transpose so it stays on the VALU copy path.
// ---------------------------------------------------------------------------
__global__ __launch_bounds__(256) void attn_kernel(
    const __bf16* __restrict__ qbuf, const __bf16* __restrict__ kbuf,
    const __bf16* __restrict__ vbuf, __bf16* __restrict__ abuf) {
  __shared__ __bf16 sQ[128][DH];     // 16 KB
  __shared__ __bf16 sK[64][DH];      //  8 KB
  __shared__ __bf16 sVt[DH][64];     //  8 KB  (transposed: [dh][seq])
  __shared__ __bf16 sP[8][16][64];   // 16 KB  (per-wave C->A relayout)

  const int rowTile = blockIdx.x;          // 0..15
  const int bh      = blockIdx.y;          // 0..31
  const int rowbase = rowTile * 128;
  const int tid = threadIdx.x, wave = tid >> 5, lane = tid & 31;
  const int l15 = lane & 15;
  const int rofs = (lane < 16) ? 0 : 8;
  const int ah  = (lane < 16) ? 0 : 16;
  const int bhf = (lane < 16) ? 0 : 16;

  const __bf16* qp = qbuf + (size_t)bh * SEQ * DH;
  const __bf16* kp = kbuf + (size_t)bh * SEQ * DH;
  const __bf16* vp = vbuf + (size_t)bh * SEQ * DH;

  // load Q tile 128x64 (async direct-to-LDS)
#pragma unroll
  for (int p = 0; p < 4; ++p) {
    int li = p * 2048 + tid * 8;
    int r = li >> 6, c = li & 63;
    copy16_to_lds(qp + (size_t)(rowbase + r) * DH + c, &sQ[r][c]);
  }
  async_wait();
  __syncthreads();

  bf16x16 qf[2];
#pragma unroll
  for (int ks = 0; ks < 2; ++ks) {
    int row = wave * 16 + l15;
    qf[ks] = mk16(*(const bf16x8*)&sQ[row][ks * 32 + ah],
                  *(const bf16x8*)&sQ[row][ks * 32 + ah + 8]);
  }

  const v8f vzero = {0.f, 0.f, 0.f, 0.f, 0.f, 0.f, 0.f, 0.f};
  v8f oacc[4];
#pragma unroll
  for (int i = 0; i < 4; ++i) oacc[i] = vzero;
  float mrow[8], lrow[8];
#pragma unroll
  for (int r = 0; r < 8; ++r) { mrow[r] = -1e30f; lrow[r] = 0.f; }

  const int njs = 2 * rowTile + 2;   // causal: j*64 covers <= rowbase+127
  for (int j = 0; j < njs; ++j) {
    __syncthreads();   // previous iter done reading sK/sVt
    // load K tile 64x64 (async direct-to-LDS)
#pragma unroll
    for (int p = 0; p < 2; ++p) {
      int li = p * 2048 + tid * 8;
      int r = li >> 6, c = li & 63;
      copy16_to_lds(kp + (size_t)(j * 64 + r) * DH + c, &sK[r][c]);
    }
    // load V tile 64x64 transposed into sVt[dh][seq]
#pragma unroll
    for (int p = 0; p < 2; ++p) {
      int li = p * 2048 + tid * 8;
      int r = li >> 6, c = li & 63;
      bf16x8 vv = *(const bf16x8*)(vp + (size_t)(j * 64 + r) * DH + c);
#pragma unroll
      for (int i = 0; i < 8; ++i) sVt[c + i][r] = vv[i];
    }
    async_wait();
    __syncthreads();

    // S = Q K^T  (16 rows x 64 cols per wave)
    v8f sacc[4];
#pragma unroll
    for (int i = 0; i < 4; ++i) sacc[i] = vzero;
#pragma unroll
    for (int ks = 0; ks < 2; ++ks) {
#pragma unroll
      for (int nt = 0; nt < 4; ++nt) {
        int col = nt * 16 + l15;
        bf16x16 kf = *(const bf16x16*)&sK[col][ks * 32 + bhf];
        sacc[nt] = wmma_bf16(qf[ks], kf, sacc[nt]);
      }
    }

    // online softmax
    float pv[4][8];
#pragma unroll
    for (int r = 0; r < 8; ++r) {
      int grow = rowbase + wave * 16 + r + rofs;
      float mc = -1e30f;
#pragma unroll
      for (int nt = 0; nt < 4; ++nt) {
        int gcol = j * 64 + nt * 16 + l15;
        float s = sacc[nt][r] * 0.125f;
        if (gcol > grow) s = -1e30f;
        pv[nt][r] = s;
        mc = fmaxf(mc, s);
      }
#pragma unroll
      for (int msk = 1; msk <= 8; msk <<= 1)
        mc = fmaxf(mc, __shfl_xor(mc, msk, 32));
      float mn = fmaxf(mrow[r], mc);
      float alpha = __expf(mrow[r] - mn);
      float lc = 0.f;
#pragma unroll
      for (int nt = 0; nt < 4; ++nt) {
        float p = __expf(pv[nt][r] - mn);
        pv[nt][r] = p;
        lc += p;
      }
#pragma unroll
      for (int msk = 1; msk <= 8; msk <<= 1) lc += __shfl_xor(lc, msk, 32);
      lrow[r] = lrow[r] * alpha + lc;
      mrow[r] = mn;
#pragma unroll
      for (int dt = 0; dt < 4; ++dt) oacc[dt][r] = oacc[dt][r] * alpha;
#pragma unroll
      for (int nt = 0; nt < 4; ++nt)
        sP[wave][r + rofs][nt * 16 + l15] = (__bf16)pv[nt][r];
    }

    // O += P V  (per-wave sP relayout; same-wave DS ops stay in order)
#pragma unroll
    for (int ks = 0; ks < 2; ++ks) {
      bf16x16 pf = mk16(*(const bf16x8*)&sP[wave][l15][ks * 32 + ah],
                        *(const bf16x8*)&sP[wave][l15][ks * 32 + ah + 8]);
#pragma unroll
      for (int dt = 0; dt < 4; ++dt) {
        bf16x16 vf = *(const bf16x16*)&sVt[dt * 16 + l15][ks * 32 + bhf];
        oacc[dt] = wmma_bf16(pf, vf, oacc[dt]);
      }
    }
  }

  // normalize + write [b, pos, h*64+dh] bf16
  const int b = bh >> 4, h = bh & 15;
#pragma unroll
  for (int r = 0; r < 8; ++r) {
    float inv = 1.0f / lrow[r];
    int pos = rowbase + wave * 16 + r + rofs;
#pragma unroll
    for (int dt = 0; dt < 4; ++dt) {
      int dh = dt * 16 + l15;
      abuf[((size_t)(b * SEQ + pos)) * DMODEL + h * DH + dh] =
          (__bf16)(oacc[dt][r] * inv);
    }
  }
}

// ---------------------------------------------------------------------------
// Kernel 3: out = attn @ w_out  (bf16 A async-staged, fp32 W converted,
// fp32 out)
// ---------------------------------------------------------------------------
__global__ __launch_bounds__(256) void outproj_kernel(
    const __bf16* __restrict__ a, const float* __restrict__ w,
    float* __restrict__ out) {
  __shared__ __bf16 sA[128][32];
  __shared__ __bf16 sB[128][32];

  const int tid  = threadIdx.x;
  const int wave = tid >> 5, lane = tid & 31;
  const int wm = wave >> 1, wn = wave & 1;
  const int tileM = blockIdx.x * 128;
  const int tileN = blockIdx.y * 128;
  const int l15 = lane & 15;
  const int ah  = (lane < 16) ? 0 : 16;
  const int bhf = (lane < 16) ? 0 : 16;

  const v8f vzero = {0.f, 0.f, 0.f, 0.f, 0.f, 0.f, 0.f, 0.f};
  v8f acc[2][4];
#pragma unroll
  for (int i = 0; i < 2; ++i)
#pragma unroll
    for (int j = 0; j < 4; ++j) acc[i][j] = vzero;

  for (int k0 = 0; k0 < DMODEL; k0 += 32) {
    // A tile: 128x32 bf16 direct (async direct-to-LDS)
#pragma unroll
    for (int p = 0; p < 2; ++p) {
      int li = p * 2048 + tid * 8;
      int r = li >> 5, c = li & 31;
      copy16_to_lds(a + (size_t)(tileM + r) * DMODEL + k0 + c, &sA[r][c]);
    }
    // B tile: 32x128 fp32 -> bf16 transposed
#pragma unroll
    for (int p = 0; p < 4; ++p) {
      int r = p * 8 + (tid >> 5);
      int c = (tid & 31) * 4;
      float4 f = *(const float4*)(w + (size_t)(k0 + r) * DMODEL + tileN + c);
      sB[c + 0][r] = (__bf16)f.x; sB[c + 1][r] = (__bf16)f.y;
      sB[c + 2][r] = (__bf16)f.z; sB[c + 3][r] = (__bf16)f.w;
    }
    async_wait();
    __syncthreads();

    bf16x16 af[2], bf[4];
#pragma unroll
    for (int mt = 0; mt < 2; ++mt) {
      int row = wm * 32 + mt * 16 + l15;
      af[mt] = mk16(*(const bf16x8*)&sA[row][ah],
                    *(const bf16x8*)&sA[row][ah + 8]);
    }
#pragma unroll
    for (int nt = 0; nt < 4; ++nt) {
      int col = wn * 64 + nt * 16 + l15;
      bf[nt] = *(const bf16x16*)&sB[col][bhf];
    }
#pragma unroll
    for (int mt = 0; mt < 2; ++mt)
#pragma unroll
      for (int nt = 0; nt < 4; ++nt)
        acc[mt][nt] = wmma_bf16(af[mt], bf[nt], acc[mt][nt]);
    __syncthreads();
  }

  const int rofs = (lane < 16) ? 0 : 8;
#pragma unroll
  for (int mt = 0; mt < 2; ++mt)
#pragma unroll
    for (int nt = 0; nt < 4; ++nt)
#pragma unroll
      for (int r = 0; r < 8; ++r) {
        int grow = tileM + wm * 32 + mt * 16 + r + rofs;
        int gcol = tileN + wn * 64 + nt * 16 + l15;
        out[(size_t)grow * DMODEL + gcol] = acc[mt][nt][r];
      }
}

extern "C" void kernel_launch(void* const* d_in, const int* in_sizes, int n_in,
                              void* d_out, int out_size, void* d_ws,
                              size_t ws_size, hipStream_t stream) {
  (void)in_sizes; (void)n_in; (void)out_size; (void)ws_size;
  const float* x     = (const float*)d_in[0];
  const float* w_qkv = (const float*)d_in[1];
  const float* w_out = (const float*)d_in[2];
  float* out = (float*)d_out;

  const size_t per_buf = (size_t)B_ * HEADS_ * SEQ * DH;  // elements
  char* ws = (char*)d_ws;
  __bf16* qbuf = (__bf16*)(ws);
  __bf16* kbuf = (__bf16*)(ws + per_buf * 2);
  __bf16* vbuf = (__bf16*)(ws + per_buf * 4);
  __bf16* abuf = (__bf16*)(ws + per_buf * 6);

  qkv_rope_kernel<<<dim3(32, 24), 256, 0, stream>>>(x, w_qkv, qbuf, kbuf, vbuf);
  attn_kernel<<<dim3(16, 32), 256, 0, stream>>>(qbuf, kbuf, vbuf, abuf);
  outproj_kernel<<<dim3(32, 8), 256, 0, stream>>>(abuf, w_out, out);
}
